// EmbeddingGenerator_47201690583126
// MI455X (gfx1250) — matrix operand order repaired and checked by reading, not compiled, not run on hardware
//
#include <hip/hip_runtime.h>
#include <hip/hip_bf16.h>

typedef __bf16 bf16;
typedef bf16  v16bf __attribute__((ext_vector_type(16)));
typedef bf16  v8bf  __attribute__((ext_vector_type(8)));
typedef float v8f   __attribute__((ext_vector_type(8)));

#define C_DIM 512
#define L_DIM 2048
#define B_DIM 16
#define LT    64              // L-columns per block
#define HALO  4               // max dilation * (K-1)/2
#define JCOLS (LT + 2*HALO)   // 72

// ---- dynamic LDS layout (bytes) ----
#define OFF_XNCL 0                          // bf16 [512][72]  conv layout (with halo)
#define SZ_XNCL  (C_DIM * JCOLS * 2)
#define OFF_XNLC (OFF_XNCL + SZ_XNCL)       // bf16 [64][512]  matmul-B layout
#define SZ_XNLC  (LT * C_DIM * 2)
#define OFF_RED  (OFF_XNLC + SZ_XNLC)       // f32 [3][72] sum(x^2) partials
#define SZ_RED   (3 * JCOLS * 4)
#define OFF_REDX (OFF_RED + SZ_RED)         // f32 [3][72] sum(x*lw) partials
#define SZ_REDX  (3 * JCOLS * 4)
#define OFF_RINV (OFF_REDX + SZ_REDX)       // f32 [72] pre-norm rsqrt
#define OFF_S1   (OFF_RINV + JCOLS * 4)     // f32 [64] sum(y*post_w*lw)
#define OFF_S2   (OFF_S1 + LT * 4)          // f32 [64] sum(y*y)
#define OFF_OUTX (OFF_S2 + LT * 4)          // f32 [64] sum(x*lw)
#define LDS_BYTES (OFF_OUTX + LT * 4)

static_assert(LDS_BYTES <= 160 * 1024, "keep <=160KB so 2 workgroups fit per 320KB WGP");

#define GW_BF16_BYTES (C_DIM * C_DIM * 2)   // 512 KB workspace for bf16 gate_w

// ---- prep kernel: gate_w fp32 -> bf16 (runs once per launch; L2-resident afterwards) ----
__global__ __launch_bounds__(256)
void convert_gw_kernel(const float* __restrict__ gw, bf16* __restrict__ gw_bf)
{
    const int i = (blockIdx.x * 256 + threadIdx.x) * 4;   // 4 elems/thread, b64 store
    float4 f = *(const float4*)(gw + i);
    v8bf h4;  // use low 4 lanes
    h4[0] = (bf16)f.x; h4[1] = (bf16)f.y; h4[2] = (bf16)f.z; h4[3] = (bf16)f.w;
    bf16* p = gw_bf + i;
    p[0] = h4[0]; p[1] = h4[1]; p[2] = h4[2]; p[3] = h4[3];
}

template<bool PRECONV>
__global__ __launch_bounds__(256)
void fused_convgate_kernel(const float* __restrict__ x,
                           const float* __restrict__ pre_w,
                           const float* __restrict__ cw0, const float* __restrict__ cb0,
                           const float* __restrict__ cw1, const float* __restrict__ cb1,
                           const float* __restrict__ cw2, const float* __restrict__ cb2,
                           const float* __restrict__ gw,  const bf16* __restrict__ gw_bf,
                           const float* __restrict__ gb,
                           const float* __restrict__ post_w,
                           const float* __restrict__ lw,
                           float* __restrict__ out)
{
    extern __shared__ char smem[];
    bf16*  xn_cl = (bf16*)(smem + OFF_XNCL);
    bf16*  xn_lc = (bf16*)(smem + OFF_XNLC);
    float* red   = (float*)(smem + OFF_RED);
    float* redx  = (float*)(smem + OFF_REDX);
    float* rinv  = (float*)(smem + OFF_RINV);
    float* S1    = (float*)(smem + OFF_S1);
    float* S2    = (float*)(smem + OFF_S2);
    float* outx  = (float*)(smem + OFF_OUTX);

    const int t  = threadIdx.x;
    const int b  = blockIdx.y;
    const int l0 = blockIdx.x * LT;

    // ---------- Phase 1: column sums: sum_c x^2 (72 cols) and sum_c x*lw (64 cols) ----------
    if (t < LT) { S1[t] = 0.f; S2[t] = 0.f; }
    if (t < 216) {                       // 3 stripes x 72 columns
        const int j = t % JCOLS;
        const int s = t / JCOLS;
        const int l = l0 - HALO + j;
        float asq = 0.f, ax = 0.f;
        if (l >= 0 && l < L_DIM) {
            const float* xp = x + (size_t)b * C_DIM * L_DIM + l;
            for (int c = s; c < C_DIM; c += 3) {
                const float v = xp[(size_t)c * L_DIM];
                asq += v * v;
                ax  += v * lw[c];
            }
        }
        red [s * JCOLS + j] = asq;
        redx[s * JCOLS + j] = ax;
    }
    __syncthreads();
    if (t < JCOLS) {
        const float ss = red[t] + red[JCOLS + t] + red[2 * JCOLS + t];
        rinv[t] = rsqrtf(ss * (1.0f / C_DIM) + 1e-5f);
        if (t >= HALO && t < HALO + LT)
            outx[t - HALO] = redx[t] + redx[JCOLS + t] + redx[2 * JCOLS + t];
    }
    __syncthreads();

    // ---------- Phase 2: x_norm -> bf16, two LDS layouts ----------
    for (int idx = t; idx < C_DIM * JCOLS; idx += 256) {
        const int c = idx / JCOLS;
        const int j = idx - c * JCOLS;
        const int l = l0 - HALO + j;
        float v = 0.f;
        if (l >= 0 && l < L_DIM) v = x[((size_t)b * C_DIM + c) * L_DIM + l];
        const float xnf = v * rinv[j] * pre_w[c];
        const bf16 h = (bf16)xnf;
        xn_cl[c * JCOLS + j] = h;
        if (j >= HALO && j < HALO + LT) xn_lc[(j - HALO) * C_DIM + c] = h;
    }
    __syncthreads();

    // ---------- Phase 3+4: gate GEMM (WMMA bf16) fused with SiLU / convs / post-norm partials ----
    const int wave  = t >> 5;
    const int lane  = t & 31;
    const int mrow  = lane & 15;   // M for A/D, N column for B/D
    const int khalf = lane >> 4;   // which K-half this lane holds

    for (int dt = 0; dt < 4; ++dt) {
        const int dtile = wave * 4 + dt;          // 8 waves x 4 = 32 d-tiles (512 channels)
        const int drow  = dtile * 16 + mrow;
        v8f acc[4] = {v8f{}, v8f{}, v8f{}, v8f{}};

        for (int kt = 0; kt < 16; ++kt) {
            // A tile: gate_w[d][c], 16x32, ISA A layout:
            // lane<16 holds K {0..7,16..23}; lane>=16 holds K {8..15,24..31}
            v16bf a;
            if constexpr (PRECONV) {
                const bf16* ap = gw_bf + (size_t)drow * C_DIM + kt * 32 + khalf * 8;
                const v8bf lo = *(const v8bf*)(ap);        // 16B aligned
                const v8bf hi = *(const v8bf*)(ap + 16);   // +32B
                #pragma unroll
                for (int i = 0; i < 8; ++i) { a[i] = lo[i]; a[i + 8] = hi[i]; }
            } else {
                const float* ap = gw + (size_t)drow * C_DIM + kt * 32 + khalf * 8;
                const float4 f0 = *(const float4*)(ap);
                const float4 f1 = *(const float4*)(ap + 4);
                const float4 f2 = *(const float4*)(ap + 16);
                const float4 f3 = *(const float4*)(ap + 20);
                a[0] = (bf16)f0.x; a[1] = (bf16)f0.y; a[2]  = (bf16)f0.z; a[3]  = (bf16)f0.w;
                a[4] = (bf16)f1.x; a[5] = (bf16)f1.y; a[6]  = (bf16)f1.z; a[7]  = (bf16)f1.w;
                a[8] = (bf16)f2.x; a[9] = (bf16)f2.y; a[10] = (bf16)f2.z; a[11] = (bf16)f2.w;
                a[12]= (bf16)f3.x; a[13]= (bf16)f3.y; a[14] = (bf16)f3.z; a[15] = (bf16)f3.w;
            }

            // B tiles: xn_lc[l][c]; lanes 0-15 K {0..7,16..23}, lanes 16-31 K {8..15,24..31}
            #pragma unroll
            for (int lt = 0; lt < 4; ++lt) {
                const bf16* bp = xn_lc + (lt * 16 + mrow) * C_DIM + kt * 32 + khalf * 8;
                const v8bf blo = *(const v8bf*)bp;
                const v8bf bhi = *(const v8bf*)(bp + 16);
                v16bf bb;
                #pragma unroll
                for (int i = 0; i < 8; ++i) { bb[i] = blo[i]; bb[i + 8] = bhi[i]; }
                acc[lt] = __builtin_amdgcn_wmma_f32_16x16x32_bf16(
                    false, a, false, bb, (short)0, acc[lt], false, false);
            }
        }

        // Epilogue on D registers: D element r -> (M = r + 8*khalf, N = mrow)
        float s1[4] = {0.f, 0.f, 0.f, 0.f};
        float s2[4] = {0.f, 0.f, 0.f, 0.f};
        #pragma unroll
        for (int r = 0; r < 8; ++r) {
            const int c = dtile * 16 + khalf * 8 + r;
            const float w00 = cw0[c*3+0], w01 = cw0[c*3+1], w02 = cw0[c*3+2];
            const float w10 = cw1[c*3+0], w11 = cw1[c*3+1], w12 = cw1[c*3+2];
            const float w20 = cw2[c*3+0], w21 = cw2[c*3+1], w22 = cw2[c*3+2];
            const float cbs  = cb0[c] + cb1[c] + cb2[c];
            const float gbc  = gb[c];
            const float pwlw = post_w[c] * lw[c];
            const bf16* xr = xn_cl + c * JCOLS;
            #pragma unroll
            for (int lt = 0; lt < 4; ++lt) {
                const int j = HALO + lt * 16 + mrow;       // xn_cl column (halo-shifted)
                float g = acc[lt][r] + gbc;
                g = g / (1.f + __expf(-g));                // SiLU
                const float xd = cbs
                    + w00 * (float)xr[j - 1] + w01 * (float)xr[j] + w02 * (float)xr[j + 1]
                    + w10 * (float)xr[j - 2] + w11 * (float)xr[j] + w12 * (float)xr[j + 2]
                    + w20 * (float)xr[j - 4] + w21 * (float)xr[j] + w22 * (float)xr[j + 4];
                const float y = xd * g;
                s1[lt] += y * pwlw;
                s2[lt] += y * y;
            }
        }
        #pragma unroll
        for (int lt = 0; lt < 4; ++lt) {
            atomicAdd(&S1[lt * 16 + mrow], s1[lt]);        // ds_add_f32
            atomicAdd(&S2[lt * 16 + mrow], s2[lt]);
        }
    }
    __syncthreads();

    // ---------- Phase 5: finalize logits ----------
    if (t < LT) {
        const float r2 = rsqrtf(S2[t] * (1.0f / C_DIM) + 1e-5f);
        out[(size_t)b * L_DIM + l0 + t] = outx[t] + r2 * S1[t];
    }
}

extern "C" void kernel_launch(void* const* d_in, const int* in_sizes, int n_in,
                              void* d_out, int out_size, void* d_ws, size_t ws_size,
                              hipStream_t stream) {
    (void)in_sizes; (void)n_in; (void)out_size;
    // setup_inputs order:
    // 0 number_log (unused), 1 x, 2 pre_w, 3 conv_w, 4 conv_b, 5 conv1_w, 6 conv1_b,
    // 7 conv2_w, 8 conv2_b, 9 gate_w, 10 gate_b, 11 post_w, 12 logits_w
    const float* x      = (const float*)d_in[1];
    const float* pre_w  = (const float*)d_in[2];
    const float* cw0    = (const float*)d_in[3];
    const float* cb0    = (const float*)d_in[4];
    const float* cw1    = (const float*)d_in[5];
    const float* cb1    = (const float*)d_in[6];
    const float* cw2    = (const float*)d_in[7];
    const float* cb2    = (const float*)d_in[8];
    const float* gw     = (const float*)d_in[9];
    const float* gb     = (const float*)d_in[10];
    const float* post_w = (const float*)d_in[11];
    const float* lw     = (const float*)d_in[12];
    float* out = (float*)d_out;

    const bool use_ws = (ws_size >= (size_t)GW_BF16_BYTES);   // constant per run -> deterministic
    dim3 grid(L_DIM / LT, B_DIM);

    if (use_ws) {
        bf16* gw_bf = (bf16*)d_ws;
        convert_gw_kernel<<<C_DIM * C_DIM / (256 * 4), 256, 0, stream>>>(gw, gw_bf);
        (void)hipFuncSetAttribute(reinterpret_cast<const void*>(&fused_convgate_kernel<true>),
                                  hipFuncAttributeMaxDynamicSharedMemorySize, LDS_BYTES);
        fused_convgate_kernel<true><<<grid, 256, LDS_BYTES, stream>>>(
            x, pre_w, cw0, cb0, cw1, cb1, cw2, cb2, gw, gw_bf, gb, post_w, lw, out);
    } else {
        (void)hipFuncSetAttribute(reinterpret_cast<const void*>(&fused_convgate_kernel<false>),
                                  hipFuncAttributeMaxDynamicSharedMemorySize, LDS_BYTES);
        fused_convgate_kernel<false><<<grid, 256, LDS_BYTES, stream>>>(
            x, pre_w, cw0, cb0, cw1, cb1, cw2, cb2, gw, (const bf16*)nullptr, gb, post_w, lw, out);
    }
}
